// GATRegressor_22522808500484
// MI455X (gfx1250) — matrix operand order repaired and checked by reading, not compiled, not run on hardware
//
#include <hip/hip_runtime.h>
#include <hip/hip_bf16.h>

// ---------------------------------------------------------------------------
// GATRegressor for gfx1250 (MI455X):
//  - dense GEMMs via v_wmma_f32_16x16x32_bf16 (wave32, 16x16 tiles), with
//    bf16 pre-conversion, fragment-order packed W, and LDS-staged A tiles
//  - edge softmax/scatter via L2 atomics (memory-bound phase)
// ---------------------------------------------------------------------------

typedef __attribute__((ext_vector_type(16))) __bf16 v16bf;
typedef __attribute__((ext_vector_type(8)))  __bf16 v8bf;   // 16 B
typedef __attribute__((ext_vector_type(4)))  __bf16 v4bf;   // 8 B
typedef __attribute__((ext_vector_type(8)))  float  v8f;

union V16 { v16bf v; v8bf h[2]; };

__device__ __forceinline__ __bf16 f2bf(float f) {
    unsigned u = __float_as_uint(f);
    unsigned r = u + 0x7FFFu + ((u >> 16) & 1u);   // round-to-nearest-even
    unsigned short hs = (unsigned short)(r >> 16);
    return __builtin_bit_cast(__bf16, hs);
}

// order-preserving encode/decode so we can do float max with atomicMax(u32)
__device__ __forceinline__ unsigned f2ord(float f) {
    unsigned u = __float_as_uint(f);
    return (u & 0x80000000u) ? ~u : (u | 0x80000000u);
}
__device__ __forceinline__ float ord2f(unsigned u) {
    unsigned v = (u & 0x80000000u) ? (u ^ 0x80000000u) : ~u;
    return __uint_as_float(v);
}

#define ORD_NEG_INF 0x007FFFFFu   // f2ord(-inf)

__device__ __forceinline__ float lrelu(float x) { return x < 0.f ? 0.2f * x : x; }
__device__ __forceinline__ float elu(float x)   { return x > 0.f ? x : __expf(x) - 1.f; }

// ---------------------------------------------------------------------------
// f32 -> bf16 row-major conversion (4 elements / thread, b128 in / b64 out)
// ---------------------------------------------------------------------------
__global__ void cvt_bf16(const float* __restrict__ src, __bf16* __restrict__ dst,
                         long n4)
{
    long i = (long)blockIdx.x * blockDim.x + threadIdx.x;
    if (i >= n4) return;
    float4 f = ((const float4*)src)[i];
    v4bf o = { f2bf(f.x), f2bf(f.y), f2bf(f.z), f2bf(f.w) };
    *(v4bf*)(dst + i * 4) = o;
}

// ---------------------------------------------------------------------------
// Pack W[128 x NOUT] (f32, row-major) into WMMA B-fragment order:
//   Wpk[((ntile*4 + kstep)*32 + lane)*16 + i] = bf16(W[(kstep*32 + (lane>>4)*16 + i)*NOUT
//                                                      + ntile*16 + (lane&15)])
// One thread per (ntile, kstep, lane) writes its 16 contiguous halves.
// ---------------------------------------------------------------------------
__global__ void pack_w(const float* __restrict__ W, __bf16* __restrict__ Wpk, int NOUT)
{
    int t = blockIdx.x * blockDim.x + threadIdx.x;
    int total = (NOUT / 16) * 4 * 32;
    if (t >= total) return;
    int lane = t & 31;
    int ks   = (t >> 5) & 3;
    int nt   = t >> 7;
    int n     = nt * 16 + (lane & 15);
    int kbase = ks * 32 + (lane >> 4) * 16;
    __bf16* __restrict__ dst = Wpk + (size_t)t * 16;
#pragma unroll
    for (int i = 0; i < 16; ++i)
        dst[i] = f2bf(W[(size_t)(kbase + i) * NOUT + n]);
}

// ---------------------------------------------------------------------------
// WMMA GEMM: out[nrows x NOUT] = A[nrows x 128] * W[128 x NOUT]
// A: bf16 row-major, staged per-block in LDS. Wpk: fragment-order bf16.
// One wave per 16-wide N tile, 16 rows per block.
// ---------------------------------------------------------------------------
template <int NOUT>
__global__ __launch_bounds__(32 * (NOUT / 16))
void gemm_bf16_wmma(const __bf16* __restrict__ A, const __bf16* __restrict__ Wpk,
                    float* __restrict__ out, int nrows)
{
    constexpr int K = 128;
    __shared__ __bf16 atile[16 * K];     // 4 KB

    const int lane = threadIdx.x & 31;
    const int wave = threadIdx.x >> 5;   // = n tile
    const int m0   = blockIdx.x * 16;

    // cooperative A-tile stage: 16 rows x 128 halves = 256 x 16B chunks
    for (int c = threadIdx.x; c < 16 * K / 8; c += blockDim.x) {
        int row = c >> 4;
        int col = (c & 15) * 8;
        int r = m0 + row; if (r >= nrows) r = nrows - 1;
        *(v8bf*)(atile + row * K + col) = *(const v8bf*)(A + (size_t)r * K + col);
    }
    __syncthreads();

    const int mrow = lane & 15;
    const int grp  = lane >> 4;          // K half select
    const v8bf* __restrict__ wpk = (const v8bf*)Wpk;

    v8f acc = {};
#pragma unroll
    for (int ks = 0; ks < 4; ++ks) {
        const int k0 = ks * 32;
        // A fragment: lane grp=0 -> K {k0..+15}; grp=1 -> K {k0+16..+31}
        V16 a;
        a.h[0] = *(const v8bf*)(atile + mrow * K + k0 + grp * 16);
        a.h[1] = *(const v8bf*)(atile + mrow * K + k0 + grp * 16 + 8);
        // B fragment: packed contiguous per (ntile, kstep, lane)
        size_t bidx = (((size_t)wave * 4 + ks) * 32 + lane) * 2;
        V16 b;
        b.h[0] = wpk[bidx];
        b.h[1] = wpk[bidx + 1];
        acc = __builtin_amdgcn_wmma_f32_16x16x32_bf16(
            false, a.v, false, b.v, (short)0, acc, false, false);
    }

    // D tile: VGPR r -> row r + 8*grp, col wave*16 + (lane&15)
    float* __restrict__ dst = out + (size_t)(m0 + 8 * grp) * NOUT
                                  + wave * 16 + (lane & 15);
    if (m0 + 16 <= nrows) {
        // full tile (always the case when nrows % 16 == 0): branch-free stores
#pragma unroll
        for (int r = 0; r < 8; ++r)
            dst[(size_t)r * NOUT] = acc[r];
    } else {
#pragma unroll
        for (int r = 0; r < 8; ++r)
            if (m0 + r + 8 * grp < nrows)
                dst[(size_t)r * NOUT] = acc[r];
    }
}

// ---------------------------------------------------------------------------
// Per-(node,head) attention projections: as = h . a_src, ad = h . a_dst
// h laid out [N, H, 32] row-major => flat index i = node*H + head
// ---------------------------------------------------------------------------
__global__ void attn_scores(const float* __restrict__ h,
                            const float* __restrict__ a_src,
                            const float* __restrict__ a_dst,
                            float* __restrict__ as_, float* __restrict__ ad_,
                            int NH, int H)
{
    int i = blockIdx.x * blockDim.x + threadIdx.x;
    if (i >= NH) return;
    int head = i % H;
    const float* __restrict__ hv = h + (size_t)i * 32;
    const float* __restrict__ sv = a_src + head * 32;
    const float* __restrict__ dv = a_dst + head * 32;
    float accs = 0.f, accd = 0.f;
#pragma unroll
    for (int j = 0; j < 32; ++j) { float v = hv[j]; accs += v * sv[j]; accd += v * dv[j]; }
    as_[i] = accs; ad_[i] = accd;
}

// ---------------------------------------------------------------------------
// Edge passes. Edge i < E comes from edge_index, else self-loop (i-E, i-E).
// ---------------------------------------------------------------------------
template <int H>
__global__ void edge_max(const float* __restrict__ as_, const float* __restrict__ ad_,
                         const int* __restrict__ ei, unsigned* __restrict__ m,
                         int E, int ET)
{
    long e = (long)blockIdx.x * blockDim.x + threadIdx.x;
    if (e >= ET) return;
    int s = (e < E) ? ei[e]     : (int)(e - E);
    int d = (e < E) ? ei[E + e] : (int)(e - E);
#pragma unroll
    for (int h = 0; h < H; ++h) {
        float x = lrelu(as_[(size_t)s * H + h] + ad_[(size_t)d * H + h]);
        atomicMax(&m[(size_t)d * H + h], f2ord(x));
    }
}

template <int H>
__global__ void edge_expsum(const float* __restrict__ as_, const float* __restrict__ ad_,
                            const int* __restrict__ ei, const unsigned* __restrict__ m,
                            float* __restrict__ ee, float* __restrict__ den,
                            int E, int ET)
{
    long e = (long)blockIdx.x * blockDim.x + threadIdx.x;
    if (e >= ET) return;
    int s = (e < E) ? ei[e]     : (int)(e - E);
    int d = (e < E) ? ei[E + e] : (int)(e - E);
#pragma unroll
    for (int h = 0; h < H; ++h) {
        float x  = lrelu(as_[(size_t)s * H + h] + ad_[(size_t)d * H + h]);
        float mx = ord2f(m[(size_t)d * H + h]);
        float ex = __expf(x - mx);
        ee[(size_t)e * H + h] = ex;
        atomicAdd(&den[(size_t)d * H + h], ex);
    }
}

// agg[dst, h, c] += h_src[src, h, c] * ee / den    (F = H*32 features/edge)
template <int H>
__global__ void edge_aggregate(const float* __restrict__ hsrc,
                               const float* __restrict__ ee,
                               const float* __restrict__ den,
                               const int* __restrict__ ei,
                               float* __restrict__ agg, int E, int ET)
{
    constexpr int F   = H * 32;
    constexpr int EPB = 256 / F;               // edges per block
    int  c     = threadIdx.x % F;
    int  local = threadIdx.x / F;
    long e = (long)blockIdx.x * EPB + local;
    if (e >= ET) return;
    int s = (e < E) ? ei[e]     : (int)(e - E);
    int d = (e < E) ? ei[E + e] : (int)(e - E);
    int head = c >> 5;
    float alpha = ee[(size_t)e * H + head] / (den[(size_t)d * H + head] + 1e-16f);
    float v = hsrc[(size_t)s * F + c] * alpha;
    atomicAdd(&agg[(size_t)d * F + c], v);
}

// ---------------------------------------------------------------------------
__global__ void bias_elu(float* __restrict__ x, const float* __restrict__ b,
                         long n, int F)
{
    long i = (long)blockIdx.x * blockDim.x + threadIdx.x;
    if (i >= n) return;
    x[i] = elu(x[i] + b[i % F]);
}

__global__ void final_head(const float* __restrict__ agg2, const float* __restrict__ b2,
                           const float* __restrict__ wl, const float* __restrict__ bl,
                           float* __restrict__ out, int n)
{
    int i = blockIdx.x * blockDim.x + threadIdx.x;
    if (i >= n) return;
    const float* __restrict__ v = agg2 + (size_t)i * 32;
    float acc = 0.f;
#pragma unroll
    for (int j = 0; j < 32; ++j) acc += elu(v[j] + b2[j]) * wl[j];
    out[i] = acc + bl[0];
}

__global__ void fill_u32(unsigned* __restrict__ p, unsigned v, long n)
{
    long i = (long)blockIdx.x * blockDim.x + threadIdx.x;
    if (i < n) p[i] = v;
}

// ---------------------------------------------------------------------------
static inline unsigned cdiv(long a, long b) { return (unsigned)((a + b - 1) / b); }

extern "C" void kernel_launch(void* const* d_in, const int* in_sizes, int n_in,
                              void* d_out, int out_size, void* d_ws, size_t ws_size,
                              hipStream_t stream)
{
    const float* x      = (const float*)d_in[0];
    const int*   ei     = (const int*)  d_in[1];
    const float* w1     = (const float*)d_in[2];
    const float* a1_src = (const float*)d_in[3];
    const float* a1_dst = (const float*)d_in[4];
    const float* b1     = (const float*)d_in[5];
    const float* w2     = (const float*)d_in[6];
    const float* a2_src = (const float*)d_in[7];
    const float* a2_dst = (const float*)d_in[8];
    const float* b2     = (const float*)d_in[9];
    const float* wl     = (const float*)d_in[10];
    const float* bl     = (const float*)d_in[11];
    float*       out    = (float*)d_out;

    const int N  = in_sizes[0] / 128;   // 100000
    const int E  = in_sizes[1] / 2;     // 1600000
    const int ET = E + N;               // + self loops

    // ---- workspace carve-out ----------------------------------------------
    float* ws   = (float*)d_ws;
    float* h1   = ws;                       // N*128 f32
    float* agg1 = h1   + (size_t)N * 128;   // N*128 f32 (layer-2 input after ELU)
    float* g2   = agg1 + (size_t)N * 128;   // N*32  f32
    float* agg2 = g2   + (size_t)N * 32;    // N*32  f32
    float* as_  = agg2 + (size_t)N * 32;    // N*4   f32
    float* ad_  = as_  + (size_t)N * 4;     // N*4   f32
    unsigned* m = (unsigned*)(ad_ + (size_t)N * 4);   // N*4 u32
    float* den  = (float*)(m + (size_t)N * 4);        // N*4 f32
    float* ee   = den  + (size_t)N * 4;     // ET*4  f32
    __bf16* xbf   = (__bf16*)(ee + (size_t)ET * 4);   // N*128 bf16
    __bf16* h2bf  = xbf  + (size_t)N * 128;           // N*128 bf16
    __bf16* w1pk  = h2bf + (size_t)N * 128;           // 128*128 bf16
    __bf16* w2pk  = w1pk + 128 * 128;                 // 128*32  bf16

    const int T = 256;

    // ---- init accumulators + bf16 staging ----------------------------------
    fill_u32<<<cdiv((long)N * 128, T), T, 0, stream>>>((unsigned*)agg1, 0u, (long)N * 128);
    fill_u32<<<cdiv((long)N * 32,  T), T, 0, stream>>>((unsigned*)agg2, 0u, (long)N * 32);
    fill_u32<<<cdiv((long)N * 4,   T), T, 0, stream>>>(m, ORD_NEG_INF, (long)N * 4);
    fill_u32<<<cdiv((long)N * 4,   T), T, 0, stream>>>((unsigned*)den, 0u, (long)N * 4);
    cvt_bf16<<<cdiv((long)N * 32, T), T, 0, stream>>>(x, xbf, (long)N * 32);
    pack_w<<<cdiv(1024, T), T, 0, stream>>>(w1, w1pk, 128);
    pack_w<<<cdiv(256,  T), T, 0, stream>>>(w2, w2pk, 32);

    // ================= layer 1 (H=4, C=32, concat) ==========================
    gemm_bf16_wmma<128><<<cdiv(N, 16), 256, 0, stream>>>(xbf, w1pk, h1, N);
    attn_scores<<<cdiv((long)N * 4, T), T, 0, stream>>>(h1, a1_src, a1_dst, as_, ad_, N * 4, 4);
    edge_max<4>   <<<cdiv(ET, T), T, 0, stream>>>(as_, ad_, ei, m, E, ET);
    edge_expsum<4><<<cdiv(ET, T), T, 0, stream>>>(as_, ad_, ei, m, ee, den, E, ET);
    edge_aggregate<4><<<cdiv(ET, 2), 256, 0, stream>>>(h1, ee, den, ei, agg1, E, ET);
    bias_elu<<<cdiv((long)N * 128, T), T, 0, stream>>>(agg1, b1, (long)N * 128, 128);

    // ================= layer 2 (H=1, C=32, mean==identity) ==================
    fill_u32<<<cdiv((long)N, T), T, 0, stream>>>(m, ORD_NEG_INF, (long)N);
    fill_u32<<<cdiv((long)N, T), T, 0, stream>>>((unsigned*)den, 0u, (long)N);
    cvt_bf16<<<cdiv((long)N * 32, T), T, 0, stream>>>(agg1, h2bf, (long)N * 32);

    gemm_bf16_wmma<32><<<cdiv(N, 16), 64, 0, stream>>>(h2bf, w2pk, g2, N);
    attn_scores<<<cdiv((long)N, T), T, 0, stream>>>(g2, a2_src, a2_dst, as_, ad_, N, 1);
    edge_max<1>   <<<cdiv(ET, T), T, 0, stream>>>(as_, ad_, ei, m, E, ET);
    edge_expsum<1><<<cdiv(ET, T), T, 0, stream>>>(as_, ad_, ei, m, ee, den, E, ET);
    edge_aggregate<1><<<cdiv(ET, 8), 256, 0, stream>>>(g2, ee, den, ei, agg2, E, ET);

    // ================= final linear head ====================================
    final_head<<<cdiv(N, T), T, 0, stream>>>(agg2, b2, wl, bl, out, N);
}
